// VQ_AE_2D_28389733826789
// MI455X (gfx1250) — matrix operand order repaired and checked by reading, not compiled, not run on hardware
//
#include <hip/hip_runtime.h>
#include <cstdint>

// ---------------- CDNA5 WMMA types ----------------
typedef __bf16 bf16x16 __attribute__((ext_vector_type(16)));
typedef float  f32x8   __attribute__((ext_vector_type(8)));
typedef int    v4i     __attribute__((ext_vector_type(4)));

union FragAB { uint4 q[2]; bf16x16 v; };

__device__ __forceinline__ unsigned short f2bfu(float f) {
    union { __bf16 b; unsigned short u; } c;
    c.b = (__bf16)f;            // native v_cvt to bf16 (RNE)
    return c.u;
}

// ---------------- async global -> LDS copy (CDNA5) ----------------
__device__ __forceinline__ void async_copy_b128(const unsigned short* g, unsigned short* l)
{
#if __has_builtin(__builtin_amdgcn_global_load_async_to_lds_b128)
    __builtin_amdgcn_global_load_async_to_lds_b128(
        (__attribute__((address_space(1))) v4i*)g,
        (__attribute__((address_space(3))) v4i*)l, 0, 0);
#else
    *(uint4*)l = *(const uint4*)g;       // fallback: synchronous copy
#endif
}

__device__ __forceinline__ void wait_async()
{
#if __has_builtin(__builtin_amdgcn_s_wait_asynccnt)
    __builtin_amdgcn_s_wait_asynccnt(0);
#elif __has_builtin(__builtin_amdgcn_global_load_async_to_lds_b128)
    asm volatile("s_wait_asynccnt 0x0" ::: "memory");
#endif
}

// =====================================================================
// bf16 WMMA GEMM (NT): C[M,N] = act( scale * A[M,K] x Bt[N,K]^T + bias ) (+resid)
// A, Bt bf16 row-major; C fp32 or bf16 (outBf16).
// Block = 128 threads (4 waves), tile 64x64, K-step 32, double-buffered
// async global->LDS staging; each wave: 2x2 v_wmma_f32_16x16x32_bf16.
// Requires M%64==0, K%32==0. OOB N-rows of Bt may be read (padded ws).
// =====================================================================
__global__ __launch_bounds__(128) void gemm_bf16_kernel(
    const unsigned short* __restrict__ A, long long sAb, long long sAh, int lda,
    const unsigned short* __restrict__ Bt, long long sBb, long long sBh, int ldb,
    const float* __restrict__ bias,
    void* __restrict__ Cv, long long sCb, long long sCh, int ldc,
    const float* __restrict__ resid,
    int M, int N, int K, float scale, int act, int outBf16, int Hdim)
{
    __shared__ unsigned short sA[2][64 * 40];   // 64 rows x 32 K (pad 40)
    __shared__ unsigned short sB[2][64 * 40];   // 64 N-rows x 32 K (pad 40)

    const int z  = blockIdx.z;
    const int bb = z / Hdim, hh = z % Hdim;
    A  += bb * sAb + hh * sAh;
    Bt += bb * sBb + hh * sBh;
    const long long cbase = bb * sCb + hh * sCh;

    const int tid  = threadIdx.x;
    const int lane = tid & 31;
    const int wave = tid >> 5;
    const int wm = (wave >> 1) * 32, wn = (wave & 1) * 32;
    const int m0 = blockIdx.y * 64,  n0 = blockIdx.x * 64;
    const int rr = lane & 15, hi = lane >> 4;
    const int kba = hi * 8;      // A frag: lanes16-31 start at K=8
    const int kbb = hi * 16;     // unused: B frag reads at kba, same k order as A

    // per-thread staging coords: 256 chunks of 8 halves (16B) per tile
    const int ar0 = tid >> 2,         ac0 = (tid & 3) * 8;
    const int ar1 = (tid + 128) >> 2, ac1 = ((tid + 128) & 3) * 8;

    auto stage = [&](int buf, int kk) {
        async_copy_b128(A  + (long long)(m0 + ar0) * lda + kk + ac0, &sA[buf][ar0 * 40 + ac0]);
        async_copy_b128(A  + (long long)(m0 + ar1) * lda + kk + ac1, &sA[buf][ar1 * 40 + ac1]);
        async_copy_b128(Bt + (long long)(n0 + ar0) * ldb + kk + ac0, &sB[buf][ar0 * 40 + ac0]);
        async_copy_b128(Bt + (long long)(n0 + ar1) * ldb + kk + ac1, &sB[buf][ar1 * 40 + ac1]);
    };

    f32x8 acc[2][2];
    #pragma unroll
    for (int i = 0; i < 2; ++i)
        #pragma unroll
        for (int j = 0; j < 2; ++j)
            #pragma unroll
            for (int r = 0; r < 8; ++r) acc[i][j][r] = 0.0f;

    const int nsteps = K >> 5;
    stage(0, 0);
    wait_async();
    __syncthreads();

    for (int s = 0; s < nsteps; ++s) {
        const int cur = s & 1;
        if (s + 1 < nsteps) stage(cur ^ 1, (s + 1) << 5);

        const unsigned short* As = sA[cur];
        const unsigned short* Bs = sB[cur];
        FragAB a[2], b[2];
        #pragma unroll
        for (int i = 0; i < 2; ++i) {
            const int arow = wm + i * 16 + rr;
            a[i].q[0] = *(const uint4*)&As[arow * 40 + kba];
            a[i].q[1] = *(const uint4*)&As[arow * 40 + 16 + kba];
        }
        #pragma unroll
        for (int j = 0; j < 2; ++j) {
            const int brow = wn + j * 16 + rr;
            b[j].q[0] = *(const uint4*)&Bs[brow * 40 + kba];
            b[j].q[1] = *(const uint4*)&Bs[brow * 40 + 16 + kba];
        }
        #pragma unroll
        for (int i = 0; i < 2; ++i)
            #pragma unroll
            for (int j = 0; j < 2; ++j)
                acc[i][j] = __builtin_amdgcn_wmma_f32_16x16x32_bf16(
                    false, a[i].v, false, b[j].v, (short)0, acc[i][j], false, false);

        wait_async();
        __syncthreads();
    }

    // ---- fused epilogue ----
    float*          Cf = (float*)Cv;
    unsigned short* Ch = (unsigned short*)Cv;
    #pragma unroll
    for (int i = 0; i < 2; ++i)
        #pragma unroll
        for (int j = 0; j < 2; ++j)
            #pragma unroll
            for (int r = 0; r < 8; ++r) {
                int row = m0 + wm + i * 16 + hi * 8 + r;
                int col = n0 + wn + j * 16 + rr;
                if (col < N) {
                    float v = acc[i][j][r] * scale;
                    if (bias)  v += bias[col];
                    if (resid) v += resid[(long long)row * ldc + col];
                    if (act == 1)      v = tanhf(v);
                    else if (act == 2) v = 0.5f * v * (1.0f + erff(v * 0.70710678118654752f));
                    long long idx = cbase + (long long)row * ldc + col;
                    if (outBf16) Ch[idx] = f2bfu(v);
                    else         Cf[idx] = v;
                }
            }
}

// ---------------- weight convert (+ transpose) fp32 -> bf16 ----------------
// in: [R,C] fp32, out: [C,R] bf16 (out[c*R+r] = in[r*C+c]); layered via z.
__global__ __launch_bounds__(256) void cvtT_kernel(const float* __restrict__ in,
                                                   unsigned short* __restrict__ out,
                                                   int R, int C)
{
    long long base = (long long)blockIdx.z * R * C;
    int i = blockIdx.x * 256 + threadIdx.x;
    if (i < R * C) {
        int c = i / R, r = i % R;
        out[base + i] = f2bfu(in[base + (long long)r * C + c]);
    }
}

__global__ __launch_bounds__(256) void cvt_kernel(const float* __restrict__ in,
                                                  unsigned short* __restrict__ out, int n)
{
    int i = blockIdx.x * 256 + threadIdx.x;
    if (i < n) out[i] = f2bfu(in[i]);
}

// ---------------- LayerNorm: fp32 in, bf16 out ----------------
__global__ __launch_bounds__(256) void layernorm_kernel(
    const float* __restrict__ x, const float* __restrict__ w,
    const float* __restrict__ b, unsigned short* __restrict__ out, int Dn)
{
    __shared__ float red[256];
    const long long row = blockIdx.x;
    const int tid = threadIdx.x;
    const float* xr = x + row * Dn;
    float s1 = 0.f, s2 = 0.f;
    for (int d = tid; d < Dn; d += 256) { float v = xr[d]; s1 += v; s2 += v * v; }
    red[tid] = s1; __syncthreads();
    for (int s = 128; s > 0; s >>= 1) { if (tid < s) red[tid] += red[tid + s]; __syncthreads(); }
    const float mean = red[0] / Dn;
    __syncthreads();
    red[tid] = s2; __syncthreads();
    for (int s = 128; s > 0; s >>= 1) { if (tid < s) red[tid] += red[tid + s]; __syncthreads(); }
    const float rstd = rsqrtf(red[0] / Dn - mean * mean + 1e-5f);
    unsigned short* orow = out + row * Dn;
    for (int d = tid; d < Dn; d += 256) orow[d] = f2bfu((xr[d] - mean) * rstd * w[d] + b[d]);
}

// ---------------- softmax (fp32 scores in, bf16 probs out) ----------------
__global__ __launch_bounds__(256) void softmax256_kernel(const float* __restrict__ att,
                                                         unsigned short* __restrict__ out)
{
    __shared__ float red[256];
    const long long row = blockIdx.x;
    const int tid = threadIdx.x;
    float v = att[row * 256 + tid];
    red[tid] = v; __syncthreads();
    for (int s = 128; s > 0; s >>= 1) { if (tid < s) red[tid] = fmaxf(red[tid], red[tid + s]); __syncthreads(); }
    const float m = red[0];
    __syncthreads();
    float e = __expf(v - m);
    red[tid] = e; __syncthreads();
    for (int s = 128; s > 0; s >>= 1) { if (tid < s) red[tid] += red[tid + s]; __syncthreads(); }
    out[row * 256 + tid] = f2bfu(e / red[0]);
}

// ---------------- misc elementwise ----------------
__global__ __launch_bounds__(256) void addpos_kernel(float* __restrict__ x,
                                                     const float* __restrict__ pos,
                                                     int total, int nd)
{
    int i = blockIdx.x * 256 + threadIdx.x;
    if (i < total) x[i] += pos[i % nd];
}

__global__ __launch_bounds__(256) void patchify_kernel(const float* __restrict__ img,
                                                       unsigned short* __restrict__ xp)
{
    int i = blockIdx.x * 256 + threadIdx.x;      // 4096*256 total
    int b = i >> 16; int rem = i & 65535;
    int n = rem >> 8, k = rem & 255;
    int ph = n >> 4, pw = n & 15, py = k >> 4, px = k & 15;
    xp[i] = f2bfu(img[(long long)b * 65536 + (ph * 16 + py) * 256 + (pw * 16 + px)]);
}

// repack V: qkv[(b*256+n)*2304 + 1536 + h*64 + dh] -> vt[((b*12+h)*64+dh)*256+n]
__global__ __launch_bounds__(256) void vrepack_kernel(const unsigned short* __restrict__ qkvb,
                                                      unsigned short* __restrict__ vt)
{
    int i = blockIdx.x * 256 + threadIdx.x;      // 3,145,728 total
    int n = i & 255; int t = i >> 8;
    int dh = t & 63; t >>= 6;
    int h = t % 12, b = t / 12;
    vt[i] = qkvb[(long long)(b * 256 + n) * 2304 + 1536 + h * 64 + dh];
}

__global__ __launch_bounds__(32) void l2norm_kernel(float* __restrict__ f)
{
    int row = blockIdx.x, lane = threadIdx.x;
    float v = f[row * 32 + lane];
    float s = v * v;
    for (int off = 16; off; off >>= 1) s += __shfl_xor(s, off, 32);
    f[row * 32 + lane] = v / fmaxf(sqrtf(s), 1e-12f);
}

__global__ void zero_losses_kernel(float* p) { p[0] = 0.f; p[1] = 0.f; }

// ---------------- VQ: argmax over 8192 codes + gather + losses ----------------
__global__ __launch_bounds__(256) void vq_kernel(
    const float* __restrict__ f, const float* __restrict__ vq,
    int* __restrict__ idxOut, unsigned short* __restrict__ bridge,
    float* lossA, float* lossB)
{
    __shared__ float fv[32];
    __shared__ float bval[256];
    __shared__ int   bidx[256];
    const int row = blockIdx.x, tid = threadIdx.x;
    if (tid < 32) fv[tid] = f[row * 32 + tid];
    __syncthreads();

    float best = -3.4e38f; int bi = 0;
    for (int c = tid; c < 8192; c += 256) {
        const float4* vp = (const float4*)(vq + c * 32);
        float dot = 0.f;
        #pragma unroll
        for (int q = 0; q < 8; ++q) {
            float4 w = vp[q];
            dot += w.x * fv[q*4+0] + w.y * fv[q*4+1] + w.z * fv[q*4+2] + w.w * fv[q*4+3];
        }
        if (dot > best) { best = dot; bi = c; }
    }
    bval[tid] = best; bidx[tid] = bi; __syncthreads();
    for (int s = 128; s > 0; s >>= 1) {
        if (tid < s) {
            float ov = bval[tid + s]; int oi = bidx[tid + s];
            if (ov > bval[tid] || (ov == bval[tid] && oi < bidx[tid])) { bval[tid] = ov; bidx[tid] = oi; }
        }
        __syncthreads();
    }
    const int idx = bidx[0];
    if (tid == 0) idxOut[row] = idx;
    if (tid < 32) {
        float e = vq[idx * 32 + tid];
        bridge[row * 32 + tid] = f2bfu(e);
        float d = e - fv[tid];
        float s = d * d;
        for (int off = 16; off; off >>= 1) s += __shfl_xor(s, off, 32);
        if (tid == 0) {
            const float inv = 1.0f / 131072.0f;
            atomicAdd(lossA, s * inv);
            atomicAdd(lossB, s * inv);
        }
    }
}

// ---------------- host-side launcher ----------------
static void gemm(hipStream_t st,
                 const unsigned short* A, long long sAb, long long sAh, int lda,
                 const unsigned short* Bt, long long sBb, long long sBh, int ldb,
                 const float* bias, void* C, long long sCb, long long sCh, int ldc,
                 const float* resid, int M, int N, int K,
                 float scale, int act, int outBf16, int Hdim, int Z)
{
    dim3 grid((N + 63) / 64, (M + 63) / 64, Z);
    gemm_bf16_kernel<<<grid, 128, 0, st>>>(A, sAb, sAh, lda, Bt, sBb, sBh, ldb,
                                           bias, C, sCb, sCh, ldc, resid,
                                           M, N, K, scale, act, outBf16, Hdim);
}

static void cvtT(hipStream_t st, const float* in, unsigned short* out, int R, int C, int L)
{
    dim3 grid((R * C + 255) / 256, 1, L);
    cvtT_kernel<<<grid, 256, 0, st>>>(in, out, R, C);
}

extern "C" void kernel_launch(void* const* d_in, const int* in_sizes, int n_in,
                              void* d_out, int out_size, void* d_ws, size_t ws_size,
                              hipStream_t stream)
{
    (void)in_sizes; (void)n_in; (void)out_size; (void)ws_size;
    auto I = [&](int i) { return (const float*)d_in[i]; };

    const float* img          = I(0);
    const float* enc_patch_w  = I(1);   // [768,256]  (already [N,K])
    const float* enc_patch_b  = I(2);
    const float* enc_pos      = I(3);
    const float* enc_ln1_w = I(4),  *enc_ln1_b = I(5),  *enc_qkv_w = I(6);
    const float* enc_proj_w = I(7), *enc_proj_b = I(8);
    const float* enc_ln2_w = I(9),  *enc_ln2_b = I(10);
    const float* enc_fc1_w = I(11), *enc_fc1_b = I(12);
    const float* enc_fc2_w = I(13), *enc_fc2_b = I(14);
    const float* enc_norm_w = I(15), *enc_norm_b = I(16);
    const float* dec_ln1_w = I(17), *dec_ln1_b = I(18), *dec_qkv_w = I(19);
    const float* dec_proj_w = I(20), *dec_proj_b = I(21);
    const float* dec_ln2_w = I(22), *dec_ln2_b = I(23);
    const float* dec_fc1_w = I(24), *dec_fc1_b = I(25);
    const float* dec_fc2_w = I(26), *dec_fc2_b = I(27);
    const float* dec_norm_w = I(28), *dec_norm_b = I(29);
    const float* in1_w = I(30), *in1_b = I(31), *in2_w = I(32), *in2_b = I(33);
    const float* vq_embed = I(34);
    const float* dec_patch_w = I(35), *dec_patch_b = I(36), *dec_pos = I(37);
    const float* out1_w = I(38), *out1_b = I(39), *out2_w = I(40), *out2_b = I(41);

    // -------- workspace layout --------
    char* wp = (char*)d_ws;
    auto allocF = [&](size_t n) { float* r = (float*)wp; wp += n * sizeof(float); return r; };
    auto allocH = [&](size_t n) { unsigned short* r = (unsigned short*)wp; wp += n * sizeof(unsigned short); return r; };

    float* x    = allocF(3145728);        // residual stream [4096,768] f32
    float* att  = allocF(12582912);       // scores [192,256,256] f32
    float* fbuf = allocF(131072);         // latent [4096,32] f32

    unsigned short* hb    = allocH(3145728);   // LN out bf16 [4096,768]
    unsigned short* qkvb  = allocH(9437184);   // [4096,2304]
    unsigned short* attb  = allocH(12582912);  // softmax out [192,256,256]
    unsigned short* bigb  = allocH(12582912);  // O / mlp / head tmp [4096,3072]
    unsigned short* vt    = allocH(3145728);   // V^T [16,12,64,256]
    unsigned short* xpb   = allocH(1048576);   // patches [4096,256]
    unsigned short* bridb = allocH(131072);    // bridge [4096,32]

    unsigned short* enc_qkvt  = allocH(12ll * 768 * 2304);
    unsigned short* enc_projt = allocH(12ll * 768 * 768);
    unsigned short* enc_fc1t  = allocH(12ll * 768 * 3072);
    unsigned short* enc_fc2t  = allocH(12ll * 3072 * 768);
    unsigned short* dec_qkvt  = allocH(3ll * 768 * 2304);
    unsigned short* dec_projt = allocH(3ll * 768 * 768);
    unsigned short* dec_fc1t  = allocH(3ll * 768 * 3072);
    unsigned short* dec_fc2t  = allocH(3ll * 3072 * 768);
    unsigned short* patchwb   = allocH(768 * 256);
    unsigned short* dpatchwb  = allocH(768 * 32);
    unsigned short* in1t      = allocH(768 * 768);
    unsigned short* in2t      = allocH(64 * 768);   // [32,768] padded to 64 rows
    unsigned short* out1t     = allocH(768 * 768);
    unsigned short* out2t     = allocH(768 * 768);

    float* dout   = (float*)d_out;       // dec_out [4096,768]
    float* loss   = dout + 3145728;      // two scalars
    int*   idxOut = (int*)(loss + 2);    // [4096]

    // -------- weight conversion (+transpose to [N,K]) --------
    cvtT(stream, enc_qkv_w,  enc_qkvt,  768, 2304, 12);
    cvtT(stream, enc_proj_w, enc_projt, 768, 768,  12);
    cvtT(stream, enc_fc1_w,  enc_fc1t,  768, 3072, 12);
    cvtT(stream, enc_fc2_w,  enc_fc2t,  3072, 768, 12);
    cvtT(stream, dec_qkv_w,  dec_qkvt,  768, 2304, 3);
    cvtT(stream, dec_proj_w, dec_projt, 768, 768,  3);
    cvtT(stream, dec_fc1_w,  dec_fc1t,  768, 3072, 3);
    cvtT(stream, dec_fc2_w,  dec_fc2t,  3072, 768, 3);
    cvtT(stream, in1_w, in1t, 768, 768, 1);
    cvtT(stream, in2_w, in2t, 768, 32,  1);
    cvtT(stream, out1_w, out1t, 768, 768, 1);
    cvtT(stream, out2_w, out2t, 768, 768, 1);
    cvt_kernel<<<(768 * 256 + 255) / 256, 256, 0, stream>>>(enc_patch_w, patchwb, 768 * 256);
    cvt_kernel<<<(768 * 32 + 255) / 256, 256, 0, stream>>>(dec_patch_w, dpatchwb, 768 * 32);

    auto run_vit = [&](const float* ln1w, const float* ln1b, const unsigned short* qkvt,
                       const unsigned short* projt, const float* projb,
                       const float* ln2w, const float* ln2b,
                       const unsigned short* f1t, const float* f1b,
                       const unsigned short* f2t, const float* f2b, int L) {
        for (int l = 0; l < L; ++l) {
            layernorm_kernel<<<4096, 256, 0, stream>>>(x, ln1w + (long long)l*768, ln1b + (long long)l*768, hb, 768);
            // qkv = h @ Wqkv (no bias) -> bf16
            gemm(stream, hb,0,0,768, qkvt + (long long)l*768*2304,0,0,768,
                 nullptr, qkvb,0,0,2304, nullptr, 4096,2304,768, 1.f,0,1, 1,1);
            // S = Q K^T * dh^-0.5 (NT, batched over 192 (b,h)) -> f32
            gemm(stream, qkvb,       256LL*2304, 64, 2304,
                         qkvb + 768, 256LL*2304, 64, 2304,
                 nullptr, att, 12LL*65536, 65536, 256, nullptr,
                 256,256,64, 0.125f,0,0, 12,192);
            softmax256_kernel<<<49152, 256, 0, stream>>>(att, attb);
            vrepack_kernel<<<12288, 256, 0, stream>>>(qkvb, vt);
            // O = att @ V  (B = V^T [64,256]) -> bf16, interleaved at col h*64
            gemm(stream, attb, 12LL*65536, 65536, 256,
                         vt,   12LL*16384, 16384, 256,
                 nullptr, bigb, 256LL*768, 64, 768, nullptr,
                 256,64,256, 1.f,0,1, 12,192);
            // x = x + O @ Wproj + b  -> f32
            gemm(stream, bigb,0,0,768, projt + (long long)l*768*768,0,0,768,
                 projb + (long long)l*768, x,0,0,768, x, 4096,768,768, 1.f,0,0, 1,1);
            layernorm_kernel<<<4096, 256, 0, stream>>>(x, ln2w + (long long)l*768, ln2b + (long long)l*768, hb, 768);
            // gelu(h @ W1 + b1) -> bf16
            gemm(stream, hb,0,0,768, f1t + (long long)l*768*3072,0,0,768,
                 f1b + (long long)l*3072, bigb,0,0,3072, nullptr, 4096,3072,768, 1.f,2,1, 1,1);
            // x = x + g @ W2 + b2 -> f32
            gemm(stream, bigb,0,0,3072, f2t + (long long)l*3072*768,0,0,3072,
                 f2b + (long long)l*768, x,0,0,768, x, 4096,768,3072, 1.f,0,0, 1,1);
        }
    };

    // ---- encoder ----
    patchify_kernel<<<4096, 256, 0, stream>>>(img, xpb);
    gemm(stream, xpb,0,0,256, patchwb,0,0,256,
         enc_patch_b, x,0,0,768, nullptr, 4096,768,256, 1.f,0,0, 1,1);
    addpos_kernel<<<12288, 256, 0, stream>>>(x, enc_pos, 3145728, 196608);
    run_vit(enc_ln1_w, enc_ln1_b, enc_qkvt, enc_projt, enc_proj_b,
            enc_ln2_w, enc_ln2_b, enc_fc1t, enc_fc1_b, enc_fc2t, enc_fc2_b, 12);
    layernorm_kernel<<<4096, 256, 0, stream>>>(x, enc_norm_w, enc_norm_b, hb, 768);

    // ---- input MLP + L2 norm ----
    gemm(stream, hb,0,0,768, in1t,0,0,768, in1_b, bigb,0,0,768, nullptr,
         4096,768,768, 1.f,1,1, 1,1);
    gemm(stream, bigb,0,0,768, in2t,0,0,768, in2_b, fbuf,0,0,32, nullptr,
         4096,32,768, 1.f,0,0, 1,1);
    l2norm_kernel<<<4096, 32, 0, stream>>>(fbuf);

    // ---- VQ ----
    zero_losses_kernel<<<1, 1, 0, stream>>>(loss);
    vq_kernel<<<4096, 256, 0, stream>>>(fbuf, vq_embed, idxOut, bridb, loss, loss + 1);

    // ---- decoder ----
    gemm(stream, bridb,0,0,32, dpatchwb,0,0,32,
         dec_patch_b, x,0,0,768, nullptr, 4096,768,32, 1.f,0,0, 1,1);
    addpos_kernel<<<12288, 256, 0, stream>>>(x, dec_pos, 3145728, 196608);
    run_vit(dec_ln1_w, dec_ln1_b, dec_qkvt, dec_projt, dec_proj_b,
            dec_ln2_w, dec_ln2_b, dec_fc1t, dec_fc1_b, dec_fc2t, dec_fc2_b, 3);
    layernorm_kernel<<<4096, 256, 0, stream>>>(x, dec_norm_w, dec_norm_b, hb, 768);

    // ---- output heads ----
    gemm(stream, hb,0,0,768, out1t,0,0,768, out1_b, bigb,0,0,768, nullptr,
         4096,768,768, 1.f,1,1, 1,1);
    gemm(stream, bigb,0,0,768, out2t,0,0,768, out2_b, dout,0,0,768, nullptr,
         4096,768,768, 1.f,0,0, 1,1);
}
